// MultiScalePropagation_39384850104343
// MI455X (gfx1250) — matrix operand order repaired and checked by reading, not compiled, not run on hardware
//
#include <hip/hip_runtime.h>
#include <cstdint>

#define BATCH 16
#define NPTS  4096
#define SPTS  46
#define KSAMP 46

typedef _Float16 v8h  __attribute__((ext_vector_type(8)));
typedef _Float16 v16h __attribute__((ext_vector_type(16)));
typedef float    v8f  __attribute__((ext_vector_type(8)));

// Global (address_space(1)) pointer types: force global_load_* (pure LOADcnt)
// instead of flat_load_* (LOADcnt+DScnt) for the GEMM fragment loads.
typedef const v8h  __attribute__((address_space(1)))* gptr_v8h;
typedef const v16h __attribute__((address_space(1)))* gptr_v16h;

// ---------------------------------------------------------------------------
// prep kernels
// ---------------------------------------------------------------------------
__global__ void k_prep_l0xyz(const float* __restrict__ xyz, float* __restrict__ l0xyz) {
  int idx = blockIdx.x * blockDim.x + threadIdx.x;
  if (idx >= BATCH * NPTS) return;
  int b = idx / NPTS, n = idx % NPTS;
  l0xyz[idx * 4 + 0] = xyz[((size_t)b * 6 + 0) * NPTS + n];
  l0xyz[idx * 4 + 1] = xyz[((size_t)b * 6 + 1) * NPTS + n];
  l0xyz[idx * 4 + 2] = xyz[((size_t)b * 6 + 2) * NPTS + n];
  l0xyz[idx * 4 + 3] = 0.f;
}

// Pack weight matrix W (Kact x Nact, row-major f32) into WMMA B-fragment order:
// fragment (nt,kt): lane l holds col nt*16+(l&15), k = kt*32 + (l>=16?16:0) + j, j=0..15.
__global__ void k_pack_w(const float* __restrict__ W, int Kact, int Nact,
                         int KT, int NT, _Float16* __restrict__ dst) {
  int idx = blockIdx.x * blockDim.x + threadIdx.x;
  int total = KT * NT * 512;
  if (idx >= total) return;
  int j = idx & 15, lane = (idx >> 4) & 31, rest = idx >> 9;
  int kt = rest % KT, nt = rest / KT;
  int k = kt * 32 + ((lane >= 16) ? 16 : 0) + j;
  int n = nt * 16 + (lane & 15);
  float v = (k < Kact && n < Nact) ? W[(size_t)k * Nact + n] : 0.f;
  dst[idx] = (_Float16)v;
}

// Fold BN: y = (acc + b)*(gm*BN_SCALE) + bt  ->  y = acc*eA + eB ; zero padding.
__global__ void k_prep_eps(const float* __restrict__ bias, const float* __restrict__ gm,
                           const float* __restrict__ bt, int Nact, int Npad,
                           float* __restrict__ eA, float* __restrict__ eB, int isConv) {
  int i = blockIdx.x * blockDim.x + threadIdx.x;
  if (i >= Npad) return;
  if (i < Nact) {
    if (isConv) { eA[i] = 1.f; eB[i] = bias[i]; }
    else {
      float s = gm[i] * rsqrtf(1.f + 1e-5f);
      eA[i] = s; eB[i] = bias[i] * s + bt[i];
    }
  } else { eA[i] = 0.f; eB[i] = 0.f; }
}

// ---------------------------------------------------------------------------
// farthest point sampling: one block per batch
// ---------------------------------------------------------------------------
__global__ void __launch_bounds__(256) k_fps(const float* __restrict__ l0xyz,
                                             float* __restrict__ distBuf,
                                             int* __restrict__ fidx) {
  __shared__ float sd[256]; __shared__ int si[256];
  __shared__ float cent[3]; __shared__ int sfar;
  int b = blockIdx.x, tid = threadIdx.x;
  const float* P = l0xyz + (size_t)b * NPTS * 4;
  float* dist = distBuf + (size_t)b * NPTS;
  for (int n = tid; n < NPTS; n += 256) dist[n] = 1e10f;
  if (tid == 0) sfar = 0;
  __syncthreads();
  for (int s = 0; s < SPTS; ++s) {
    int far = sfar;
    if (tid == 0) {
      fidx[b * SPTS + s] = far;
      cent[0] = P[far * 4]; cent[1] = P[far * 4 + 1]; cent[2] = P[far * 4 + 2];
    }
    __syncthreads();
    float cx = cent[0], cy = cent[1], cz = cent[2];
    float bd = -1.f; int bi = 0;
    for (int n = tid; n < NPTS; n += 256) {
      float dx = P[n * 4] - cx, dy = P[n * 4 + 1] - cy, dz = P[n * 4 + 2] - cz;
      float d = fminf(dist[n], dx * dx + dy * dy + dz * dz);
      dist[n] = d;
      if (d > bd) { bd = d; bi = n; }
    }
    sd[tid] = bd; si[tid] = bi;
    __syncthreads();
    for (int off = 128; off > 0; off >>= 1) {
      if (tid < off) {
        if (sd[tid + off] > sd[tid] ||
            (sd[tid + off] == sd[tid] && si[tid + off] < si[tid])) {
          sd[tid] = sd[tid + off]; si[tid] = si[tid + off];
        }
      }
      __syncthreads();
    }
    if (tid == 0) sfar = si[0];
    __syncthreads();
  }
}

__global__ void k_gather_l1(const float* __restrict__ l0xyz, const int* __restrict__ fidx,
                            float* __restrict__ l1xyz) {
  int idx = blockIdx.x * blockDim.x + threadIdx.x;
  if (idx >= BATCH * SPTS) return;
  int b = idx / SPTS;
  int f = fidx[idx];
  l1xyz[idx * 4 + 0] = l0xyz[((size_t)b * NPTS + f) * 4 + 0];
  l1xyz[idx * 4 + 1] = l0xyz[((size_t)b * NPTS + f) * 4 + 1];
  l1xyz[idx * 4 + 2] = l0xyz[((size_t)b * NPTS + f) * 4 + 2];
  l1xyz[idx * 4 + 3] = 0.f;
}

// ---------------------------------------------------------------------------
// ball query: one wave per (b,s); selects first K in-radius indices in
// ascending order (matches jnp sort-of-masked-arange semantics).
// ---------------------------------------------------------------------------
__global__ void __launch_bounds__(32) k_ball(const float* __restrict__ l0xyz,
                                             const float* __restrict__ l1xyz,
                                             int* __restrict__ gi, float r2) {
  int bs = blockIdx.x;
  int b = bs / SPTS;
  int lane = threadIdx.x;
  float cx = l1xyz[bs * 4 + 0], cy = l1xyz[bs * 4 + 1], cz = l1xyz[bs * 4 + 2];
  const float* P = l0xyz + (size_t)b * NPTS * 4;
  int* out = gi + (size_t)bs * KSAMP;
  int cnt = 0, firstIdx = -1;
  for (int base = 0; base < NPTS && cnt < KSAMP; base += 32) {
    int n = base + lane;
    float dx = P[n * 4] - cx, dy = P[n * 4 + 1] - cy, dz = P[n * 4 + 2] - cz;
    float d2 = dx * dx + dy * dy + dz * dz;
    bool in = d2 <= r2;
    unsigned mask = __builtin_amdgcn_ballot_w32(in);
    if (firstIdx < 0 && mask) {
      int fl = __ffs((int)mask) - 1;
      firstIdx = __shfl(n, fl, 32);
    }
    if (in) {
      int slot = cnt + __popc(mask & ((1u << lane) - 1u));
      if (slot < KSAMP) out[slot] = n;
    }
    cnt += __popc(mask);
  }
  if (cnt > KSAMP) cnt = KSAMP;
  if (firstIdx < 0) firstIdx = 0;
  for (int slot = cnt + lane; slot < KSAMP; slot += 32) out[slot] = firstIdx;
}

// grouped features: [xyz_cl(6) | es(32) | gx(3) | 0-pad..63] as f16 rows
__global__ void k_group_gather(const float* __restrict__ xyz, const float* __restrict__ es,
                               const float* __restrict__ l0xyz, const float* __restrict__ l1xyz,
                               const int* __restrict__ gi, _Float16* __restrict__ out) {
  int row = blockIdx.x * blockDim.x + threadIdx.x;
  if (row >= BATCH * SPTS * KSAMP) return;
  int bs = row / KSAMP;
  int b = bs / SPTS;
  int gidx = gi[row];
  _Float16* o = out + (size_t)row * 64;
  for (int c = 0; c < 6; c++)  o[c]      = (_Float16)xyz[((size_t)b * 6 + c) * NPTS + gidx];
  for (int c = 0; c < 32; c++) o[6 + c]  = (_Float16)es[((size_t)b * 32 + c) * NPTS + gidx];
  for (int c = 0; c < 3; c++)
    o[38 + c] = (_Float16)(l0xyz[((size_t)b * NPTS + gidx) * 4 + c] - l1xyz[(size_t)bs * 4 + c]);
  for (int c = 41; c < 64; c++) o[c] = (_Float16)0.f;
}

__global__ void k_maxpool_k(const _Float16* __restrict__ O, int ldO, int c2, int chOff,
                            _Float16* __restrict__ l1pts) {
  int idx = blockIdx.x * blockDim.x + threadIdx.x;
  if (idx >= BATCH * SPTS * c2) return;
  int bs = idx / c2, c = idx % c2;
  float m = -1e30f;
  for (int k = 0; k < KSAMP; k++)
    m = fmaxf(m, (float)O[((size_t)bs * KSAMP + k) * ldO + c]);
  l1pts[(size_t)bs * 640 + chOff + c] = (_Float16)m;
}

__global__ void k_build_sa3in(const float* __restrict__ l1xyz, const _Float16* __restrict__ l1pts,
                              _Float16* __restrict__ out) {
  int idx = blockIdx.x * blockDim.x + threadIdx.x;
  if (idx >= BATCH * SPTS * 672) return;
  int bs = idx / 672, c = idx % 672;
  _Float16 v;
  if (c < 3)        v = (_Float16)l1xyz[bs * 4 + c];
  else if (c < 643) v = l1pts[(size_t)bs * 640 + (c - 3)];
  else              v = (_Float16)0.f;
  out[idx] = v;
}

__global__ void k_pool_s(const _Float16* __restrict__ sa3O, _Float16* __restrict__ l3) {
  int idx = blockIdx.x * blockDim.x + threadIdx.x;
  if (idx >= BATCH * 1024) return;
  int b = idx / 1024, c = idx % 1024;
  float m = -1e30f;
  for (int s = 0; s < SPTS; s++)
    m = fmaxf(m, (float)sa3O[(size_t)(b * SPTS + s) * 1024 + c]);
  l3[idx] = (_Float16)m;
}

__global__ void k_build_fp3in(const _Float16* __restrict__ l1pts, const _Float16* __restrict__ l3,
                              _Float16* __restrict__ out) {
  int idx = blockIdx.x * blockDim.x + threadIdx.x;
  if (idx >= BATCH * SPTS * 1664) return;
  int bs = idx / 1664, c = idx % 1664;
  out[idx] = (c < 640) ? l1pts[(size_t)bs * 640 + c]
                       : l3[(size_t)(bs / SPTS) * 1024 + (c - 640)];
}

__global__ void k_knn3(const float* __restrict__ l0xyz, const float* __restrict__ l1xyz,
                       int* __restrict__ ni, float* __restrict__ wB) {
  int idx = blockIdx.x * blockDim.x + threadIdx.x;
  if (idx >= BATCH * NPTS) return;
  int b = idx / NPTS;
  float px = l0xyz[idx * 4], py = l0xyz[idx * 4 + 1], pz = l0xyz[idx * 4 + 2];
  float d0 = 1e30f, d1 = 1e30f, d2 = 1e30f; int i0 = 0, i1 = 0, i2 = 0;
  for (int s = 0; s < SPTS; s++) {
    const float* q = l1xyz + ((size_t)b * SPTS + s) * 4;
    float dx = px - q[0], dy = py - q[1], dz = pz - q[2];
    float d = dx * dx + dy * dy + dz * dz;
    if (d < d0)      { d2 = d1; i2 = i1; d1 = d0; i1 = i0; d0 = d; i0 = s; }
    else if (d < d1) { d2 = d1; i2 = i1; d1 = d;  i1 = s; }
    else if (d < d2) { d2 = d;  i2 = s; }
  }
  float w0 = 1.f / (d0 + 1e-8f), w1 = 1.f / (d1 + 1e-8f), w2 = 1.f / (d2 + 1e-8f);
  float ws = w0 + w1 + w2;
  ni[idx * 4] = i0; ni[idx * 4 + 1] = i1; ni[idx * 4 + 2] = i2; ni[idx * 4 + 3] = 0;
  wB[idx * 4] = w0 / ws; wB[idx * 4 + 1] = w1 / ws; wB[idx * 4 + 2] = w2 / ws; wB[idx * 4 + 3] = 0.f;
}

// fp1 input row: [g(512) | l0_xyz(3) | xyz_cl(6) | interp(512) | 0..1055]
__global__ void k_build_fp1in(const float* __restrict__ g, const float* __restrict__ l0xyz,
                              const float* __restrict__ xyz, const _Float16* __restrict__ l2,
                              const int* __restrict__ ni, const float* __restrict__ wB,
                              _Float16* __restrict__ out) {
  size_t idx = (size_t)blockIdx.x * blockDim.x + threadIdx.x;
  size_t total = (size_t)BATCH * NPTS * 1056;
  if (idx >= total) return;
  int row = (int)(idx / 1056);
  int c = (int)(idx % 1056);
  int b = row / NPTS, n = row % NPTS;
  float v;
  if (c < 512)      v = g[(size_t)b * 512 + c];
  else if (c < 515) v = l0xyz[(size_t)row * 4 + (c - 512)];
  else if (c < 521) v = xyz[((size_t)b * 6 + (c - 515)) * NPTS + n];
  else if (c < 1033) {
    int cc = c - 521;
    float acc = 0.f;
    for (int j = 0; j < 3; j++) {
      int s = ni[(size_t)row * 4 + j];
      acc += wB[(size_t)row * 4 + j] * (float)l2[(size_t)(b * SPTS + s) * 512 + cc];
    }
    v = acc;
  } else v = 0.f;
  out[idx] = (_Float16)v;
}

// ---------------------------------------------------------------------------
// WMMA GEMM, specialized on number of N-tiles per wave (TN): one wave
// computes a 32 x (TN*16) output tile (2 M-tiles).  Straight-line K loop,
// incrementing fragment pointers, explicit global-address-space loads so the
// backend emits global_load_b128 (LOADcnt only) rather than flat_load_b128.
// Epilogue: v = acc*eA[col]+eB[col], optional relu, optional f16 row-major
// store and/or f32 transposed (B,ncolsF,rowsPer) store.
// ---------------------------------------------------------------------------
template <int TN>
__global__ void __launch_bounds__(32) k_gemm_wmma(
    const _Float16* __restrict__ A, int lda,
    const _Float16* __restrict__ Bp, int KT, int ntOff,
    const float* __restrict__ eA, const float* __restrict__ eB, int relu,
    _Float16* __restrict__ outH, int ldo,
    float* __restrict__ outF, int ncolsF, int rowsPer) {
  const int lane = threadIdx.x;
  const int mt0 = blockIdx.x * 2;
  const int nt0 = ntOff + blockIdx.y * TN;

  v8f zero = {0.f, 0.f, 0.f, 0.f, 0.f, 0.f, 0.f, 0.f};
  v8f acc[2][TN];
#pragma unroll
  for (int mi = 0; mi < 2; mi++)
#pragma unroll
    for (int t = 0; t < TN; t++) acc[mi][t] = zero;

  const int aoff = (lane >= 16) ? 16 : 0;
  const size_t rowA0 = (size_t)(mt0 * 16 + (lane & 15));
  const _Float16* pa0 = A + rowA0 * lda + aoff;
  const _Float16* pa1 = A + (rowA0 + 16) * lda + aoff;

  const _Float16* pb[TN];
#pragma unroll
  for (int t = 0; t < TN; t++)
    pb[t] = Bp + ((size_t)(nt0 + t) * KT * 32 + lane) * 16;

  for (int kt = 0; kt < KT; ++kt) {
    // issue all loads first (global address space -> global_load_b128)
    v8h a0lo = *(gptr_v8h)(uintptr_t)(pa0);
    v8h a0hi = *(gptr_v8h)(uintptr_t)(pa0 + 8);
    v8h a1lo = *(gptr_v8h)(uintptr_t)(pa1);
    v8h a1hi = *(gptr_v8h)(uintptr_t)(pa1 + 8);
    v16h bf[TN];
#pragma unroll
    for (int t = 0; t < TN; t++) {
      bf[t] = *(gptr_v16h)(uintptr_t)(pb[t]);
      pb[t] += 512;
    }
    __builtin_prefetch((const void*)(pa0 + 32), 0, 3);
    __builtin_prefetch((const void*)(pa1 + 32), 0, 3);
    pa0 += 32; pa1 += 32;

    v16h a0, a1;
#pragma unroll
    for (int i = 0; i < 8; i++) {
      a0[i] = a0lo[i]; a0[i + 8] = a0hi[i];
      a1[i] = a1lo[i]; a1[i + 8] = a1hi[i];
    }
#pragma unroll
    for (int t = 0; t < TN; t++) {
      acc[0][t] = __builtin_amdgcn_wmma_f32_16x16x32_f16(
          false, a0, false, bf[t], (short)0, acc[0][t], false, false);
      acc[1][t] = __builtin_amdgcn_wmma_f32_16x16x32_f16(
          false, a1, false, bf[t], (short)0, acc[1][t], false, false);
    }
  }

  const int nl = lane & 15;
  const int moff = (lane >= 16) ? 8 : 0;
#pragma unroll
  for (int t = 0; t < TN; t++) {
    int col = (nt0 + t) * 16 + nl;
    float ea = eA[col], ebv = eB[col];
#pragma unroll
    for (int mi = 0; mi < 2; mi++) {
      int rbase = (mt0 + mi) * 16 + moff;
#pragma unroll
      for (int r = 0; r < 8; r++) {
        float v = acc[mi][t][r] * ea + ebv;
        if (relu) v = fmaxf(v, 0.f);
        int row = rbase + r;
        if (outH) outH[(size_t)row * ldo + col] = (_Float16)v;
        if (outF && col < ncolsF) {
          int bb = row / rowsPer, nn = row % rowsPer;
          outF[((size_t)bb * ncolsF + col) * rowsPer + nn] = v;
        }
      }
    }
  }
}

// ---------------------------------------------------------------------------
extern "C" void kernel_launch(void* const* d_in, const int* in_sizes, int n_in,
                              void* d_out, int out_size, void* d_ws, size_t ws_size,
                              hipStream_t stream) {
  (void)in_sizes; (void)n_in; (void)out_size; (void)ws_size;
  // Assumed jax-tree (sorted dict key) flatten order:
  // 0:es 1:g 2:conv1.W 3:conv1.b 4..11:fp1(L0 W,b,gm,bt; L1 W,b,gm,bt)
  // 12..19:fp3  20..59:sa1 (5 scales x 2 layers x 4)  60..67:sa3  68:xyz
  const float* es   = (const float*)d_in[0];
  const float* g    = (const float*)d_in[1];
  const float* xyz  = (const float*)d_in[68];

  size_t off = 0;
  auto alloc = [&](size_t bytes) -> void* {
    void* p = (char*)d_ws + off;
    off = (off + bytes + 255) & ~(size_t)255;
    return p;
  };

  // layers: 0-9 sa1 (s0L0,s0L1,...,s4L1) 10-11 sa3, 12-13 fp3, 14-15 fp1, 16 conv
  const int LAY = 17;
  const int wIdx[LAY] = {20,24,28,32,36,40,44,48,52,56, 60,64, 12,16, 4,8, 2};
  const int Kact[LAY] = {41,96,41,192,41,96,41,56,41,40, 643,400, 1664,1024, 1033,512, 256};
  const int Nact[LAY] = {96,128,192,256,96,128,56,64,40,64, 400,1024, 1024,512, 512,256, 32};
  const int Kpad[LAY] = {64,96,64,192,64,96,64,64,64,64, 672,416, 1664,1024, 1056,512, 256};
  const int Npad[LAY] = {96,128,192,256,96,128,64,64,64,64, 416,1024, 1024,512, 512,256, 32};

  _Float16* Wp[LAY]; float* eAp[LAY]; float* eBp[LAY];
  for (int i = 0; i < LAY; i++) {
    Wp[i]  = (_Float16*)alloc((size_t)Kpad[i] * Npad[i] * sizeof(_Float16));
    eAp[i] = (float*)alloc((size_t)Npad[i] * sizeof(float));
    eBp[i] = (float*)alloc((size_t)Npad[i] * sizeof(float));
  }

  const int GR = BATCH * SPTS * KSAMP;     // 33856 grouped rows
  const int RS = BATCH * SPTS;             // 736 sampled rows
  float* dist   = (float*)alloc((size_t)BATCH * NPTS * 4);
  float* l0xyz  = (float*)alloc((size_t)BATCH * NPTS * 4 * 4);
  int*   fidx   = (int*)alloc((size_t)BATCH * SPTS * 4);
  float* l1xyz  = (float*)alloc((size_t)BATCH * SPTS * 4 * 4);
  int*   gi     = (int*)alloc((size_t)GR * 4);
  int*   ni     = (int*)alloc((size_t)BATCH * NPTS * 4 * 4);
  float* wB     = (float*)alloc((size_t)BATCH * NPTS * 4 * 4);
  _Float16* gIn   = (_Float16*)alloc((size_t)GR * 64 * 2);
  _Float16* gH    = (_Float16*)alloc((size_t)GR * 192 * 2);
  _Float16* gO    = (_Float16*)alloc((size_t)GR * 256 * 2);
  _Float16* l1pts = (_Float16*)alloc((size_t)RS * 640 * 2);
  _Float16* sa3in = (_Float16*)alloc((size_t)RS * 672 * 2);
  _Float16* sa3H  = (_Float16*)alloc((size_t)RS * 416 * 2);
  _Float16* sa3O  = (_Float16*)alloc((size_t)RS * 1024 * 2);
  _Float16* l3    = (_Float16*)alloc((size_t)BATCH * 1024 * 2);
  _Float16* fp3in = (_Float16*)alloc((size_t)RS * 1664 * 2);
  _Float16* fp3H  = (_Float16*)alloc((size_t)RS * 1024 * 2);
  _Float16* l2    = (_Float16*)alloc((size_t)RS * 512 * 2);
  _Float16* fp1in = (_Float16*)alloc((size_t)BATCH * NPTS * 1056 * 2);
  _Float16* fp1H  = (_Float16*)alloc((size_t)BATCH * NPTS * 512 * 2);
  _Float16* l0h   = (_Float16*)alloc((size_t)BATCH * NPTS * 256 * 2);

  auto launch_gemm = [&](const _Float16* Aa, int lda, int L, int Mtiles,
                         int relu, _Float16* oH, int ldo,
                         float* oF, int ncF, int rp) {
    int KT = Kpad[L] / 32, NT = Npad[L] / 16;
    int main4 = NT / 4, rem = NT % 4;
    if (main4 > 0) {
      dim3 grd(Mtiles / 2, main4);
      k_gemm_wmma<4><<<grd, 32, 0, stream>>>(Aa, lda, Wp[L], KT, 0, eAp[L], eBp[L],
                                             relu, oH, ldo, oF, ncF, rp);
    }
    if (rem) {
      dim3 grd(Mtiles / 2, 1);
      int ntOff = main4 * 4;
      switch (rem) {
        case 1:
          k_gemm_wmma<1><<<grd, 32, 0, stream>>>(Aa, lda, Wp[L], KT, ntOff, eAp[L], eBp[L],
                                                 relu, oH, ldo, oF, ncF, rp);
          break;
        case 2:
          k_gemm_wmma<2><<<grd, 32, 0, stream>>>(Aa, lda, Wp[L], KT, ntOff, eAp[L], eBp[L],
                                                 relu, oH, ldo, oF, ncF, rp);
          break;
        default:
          k_gemm_wmma<3><<<grd, 32, 0, stream>>>(Aa, lda, Wp[L], KT, ntOff, eAp[L], eBp[L],
                                                 relu, oH, ldo, oF, ncF, rp);
          break;
      }
    }
  };

  // weight packing + epilogue folding
  for (int i = 0; i < LAY; i++) {
    int KT = Kpad[i] / 32, NT = Npad[i] / 16;
    int tot = KT * NT * 512;
    k_pack_w<<<(tot + 255) / 256, 256, 0, stream>>>(
        (const float*)d_in[wIdx[i]], Kact[i], Nact[i], KT, NT, Wp[i]);
    int isConv = (i == 16);
    const float* bptr = (const float*)d_in[wIdx[i] + 1];
    const float* gmp = isConv ? bptr : (const float*)d_in[wIdx[i] + 2];
    const float* btp = isConv ? bptr : (const float*)d_in[wIdx[i] + 3];
    k_prep_eps<<<(Npad[i] + 255) / 256, 256, 0, stream>>>(
        bptr, gmp, btp, Nact[i], Npad[i], eAp[i], eBp[i], isConv);
  }

  // geometry
  k_prep_l0xyz<<<(BATCH * NPTS + 255) / 256, 256, 0, stream>>>(xyz, l0xyz);
  k_fps<<<BATCH, 256, 0, stream>>>(l0xyz, dist, fidx);
  k_gather_l1<<<(RS + 255) / 256, 256, 0, stream>>>(l0xyz, fidx, l1xyz);

  // multi-scale grouped MLPs
  const float radii[5] = {0.1f, 0.2f, 0.3f, 0.6f, 1.0f};
  const int c1p[5] = {96, 192, 96, 64, 64};
  const int c2a[5] = {128, 256, 128, 64, 64};
  const int c2p[5] = {128, 256, 128, 64, 64};
  const int chOff[5] = {0, 128, 384, 512, 576};
  const int MtG = GR / 16;  // 2116
  for (int s5 = 0; s5 < 5; s5++) {
    k_ball<<<RS, 32, 0, stream>>>(l0xyz, l1xyz, gi, radii[s5] * radii[s5]);
    k_group_gather<<<(GR + 255) / 256, 256, 0, stream>>>(xyz, es, l0xyz, l1xyz, gi, gIn);
    launch_gemm(gIn, 64, s5 * 2, MtG, 1, gH, c1p[s5], nullptr, 1, 1);
    launch_gemm(gH, c1p[s5], s5 * 2 + 1, MtG, 1, gO, c2p[s5], nullptr, 1, 1);
    k_maxpool_k<<<(RS * c2a[s5] + 255) / 256, 256, 0, stream>>>(gO, c2p[s5], c2a[s5], chOff[s5], l1pts);
  }

  // sa3 -> l3
  k_build_sa3in<<<(RS * 672 + 255) / 256, 256, 0, stream>>>(l1xyz, l1pts, sa3in);
  launch_gemm(sa3in, 672, 10, RS / 16, 1, sa3H, 416, nullptr, 1, 1);
  launch_gemm(sa3H, 416, 11, RS / 16, 1, sa3O, 1024, nullptr, 1, 1);
  k_pool_s<<<(BATCH * 1024 + 255) / 256, 256, 0, stream>>>(sa3O, l3);

  // fp3 -> l2
  k_build_fp3in<<<(RS * 1664 + 255) / 256, 256, 0, stream>>>(l1pts, l3, fp3in);
  launch_gemm(fp3in, 1664, 12, RS / 16, 1, fp3H, 1024, nullptr, 1, 1);
  launch_gemm(fp3H, 1024, 13, RS / 16, 1, l2, 512, nullptr, 1, 1);

  // 3-NN interpolation + fp1 input
  k_knn3<<<(BATCH * NPTS + 255) / 256, 256, 0, stream>>>(l0xyz, l1xyz, ni, wB);
  {
    size_t tot = (size_t)BATCH * NPTS * 1056;
    k_build_fp1in<<<(unsigned)((tot + 255) / 256), 256, 0, stream>>>(g, l0xyz, xyz, l2, ni, wB, fp1in);
  }

  // fp1 (dominant GEMMs) + conv1, epilogues write transposed f32 into d_out
  const int MtN = (BATCH * NPTS) / 16;  // 4096
  float* out0 = (float*)d_out;                                   // (B,256,N)
  float* out1 = (float*)d_out + (size_t)BATCH * 256 * NPTS;      // (B,32,N)
  launch_gemm(fp1in, 1056, 14, MtN, 1, fp1H, 512, nullptr, 1, 1);
  launch_gemm(fp1H, 512, 15, MtN, 1, l0h, 256, out0, 256, NPTS);
  launch_gemm(l0h, 256, 16, MtN, 0, nullptr, 0, out1, 32, NPTS);
}
